// WindowAttentionCross_78520592105799
// MI455X (gfx1250) — matrix operand order repaired and checked by reading, not compiled, not run on hardware
//
#include <hip/hip_runtime.h>
#include <hip/hip_bf16.h>

typedef __attribute__((ext_vector_type(16))) _Float16 v16h;
typedef __attribute__((ext_vector_type(8)))  _Float16 v8h;
typedef __attribute__((ext_vector_type(8)))  float    v8f;
typedef __attribute__((ext_vector_type(4)))  unsigned int u32x4;
typedef __attribute__((ext_vector_type(8)))  int          i32x8;
typedef __attribute__((ext_vector_type(4)))  int          i32x4;

#if defined(__has_builtin)
#if __has_builtin(__builtin_amdgcn_tensor_load_to_lds) && \
    __has_builtin(__builtin_amdgcn_s_wait_tensorcnt)
#define HAVE_TDM 1
#endif
#endif
#ifndef HAVE_TDM
#define HAVE_TDM 0
#endif

// ---------------------------------------------------------------------------
// WMMA helpers (CDNA5 f16 -> f32, 16x16x32)
// ---------------------------------------------------------------------------
__device__ inline v8f wmma_f16(v16h a, v16h b, v8f c) {
  return __builtin_amdgcn_wmma_f32_16x16x32_f16(false, a, false, b, (short)0, c,
                                                false, false);
}

// A-fragment (16x32 f16): lane<16 row m=lane, K contiguous from kBase;
// caller passes p = row_base + kBase (kBase = 16 for lanes >= 16)
__device__ inline v16h load_a_f16(const _Float16* p) {
  v16h a;
#pragma unroll
  for (int e = 0; e < 8; ++e)  a[e] = p[e];
#pragma unroll
  for (int e = 8; e < 16; ++e) a[e] = p[e];
  return a;
}
__device__ inline v16h load_a_f32(const float* p) {
  v16h a;
#pragma unroll
  for (int e = 0; e < 8; ++e)  a[e] = (_Float16)p[e];
#pragma unroll
  for (int e = 8; e < 16; ++e) a[e] = (_Float16)p[e];
  return a;
}
// B-fragment (32x16 f16): lane holds column n=lane&15, K contiguous from kBase;
// caller passes p = col_row_base + kBase (kBase = 16 for lanes >= 16)
__device__ inline v16h load_b_f16(const _Float16* p) {
  v16h b;
#pragma unroll
  for (int e = 0; e < 16; ++e) b[e] = p[e];
  return b;
}

// token l (0..16383) on 128x128 grid -> (window 0..63, token-in-window 0..255)
__device__ inline void tok2win(int l, int& w, int& t) {
  int r = l >> 7, c = l & 127;
  w = ((r >> 4) << 3) | (c >> 4);
  t = ((r & 15) << 4) | (c & 15);
}

// ---------------------------------------------------------------------------
// weight f32 [K][N] -> f16 transposed [N][K]
// ---------------------------------------------------------------------------
__global__ void k_wtrans(const float* __restrict__ W, _Float16* __restrict__ WT,
                         int K, int N) {
  int idx = blockIdx.x * blockDim.x + threadIdx.x;
  if (idx >= K * N) return;
  int n = idx / K, k = idx - n * K;
  WT[idx] = (_Float16)W[(size_t)k * N + n];
}

// ---------------------------------------------------------------------------
// qp = q @ Wq -> windowed qw[b][w][h][t][d] (f16)
// one wave per 16x64 strip (4 N-tiles share the A fragment); M=65536,N=512,K=512
// ---------------------------------------------------------------------------
__global__ void __launch_bounds__(128)
k_proj_q(const float* __restrict__ X, const _Float16* __restrict__ WT,
         _Float16* __restrict__ qw) {
  const int wid  = blockIdx.x * (blockDim.x >> 5) + (threadIdx.x >> 5);
  const int lane = threadIdx.x & 31;
  const int lh   = lane >> 4;
  const int ng   = wid & 7;            // 8 groups of 4 N-tiles
  const int mt   = wid >> 3;           // 4096 M-tiles
  const float*    arow  = X  + (size_t)(mt * 16 + (lane & 15)) * 512 + lh * 16;
  const _Float16* brow0 = WT + (size_t)(ng * 64 + (lane & 15)) * 512 + lh * 16;
  v8f acc[4] = {};
#pragma unroll
  for (int kc = 0; kc < 16; ++kc) {
    v16h a = load_a_f32(arow + kc * 32);
#pragma unroll
    for (int u = 0; u < 4; ++u)
      acc[u] = wmma_f16(a, load_b_f16(brow0 + (size_t)u * 16 * 512 + kc * 32), acc[u]);
  }
#pragma unroll
  for (int u = 0; u < 4; ++u) {
    const int coln = ng * 64 + u * 16 + (lane & 15);
    const int h = coln >> 6, d = coln & 63;
#pragma unroll
    for (int j = 0; j < 8; ++j) {
      int M = mt * 16 + j + lh * 8;
      int bb = M >> 14, l = M & 16383, w, t;
      tok2win(l, w, t);
      qw[((((size_t)bb * 64 + w) * 8 + h) * 256 + t) * 64 + d] = (_Float16)acc[u][j];
    }
  }
}

// ---------------------------------------------------------------------------
// kvp = kv @ Wkv ; cols 0..511 -> kw[b][w][h][t][d], 512..1023 -> vt[b][w][h][d][t]
// ---------------------------------------------------------------------------
__global__ void __launch_bounds__(128)
k_proj_kv(const float* __restrict__ X, const _Float16* __restrict__ WT,
          _Float16* __restrict__ kw, _Float16* __restrict__ vt) {
  const int wid  = blockIdx.x * (blockDim.x >> 5) + (threadIdx.x >> 5);
  const int lane = threadIdx.x & 31;
  const int lh   = lane >> 4;
  const int ng   = wid & 15;           // 16 groups of 4 N-tiles (N = 1024)
  const int mt   = wid >> 4;           // 4096 M-tiles
  const float*    arow  = X  + (size_t)(mt * 16 + (lane & 15)) * 512 + lh * 16;
  const _Float16* brow0 = WT + (size_t)(ng * 64 + (lane & 15)) * 512 + lh * 16;
  v8f acc[4] = {};
#pragma unroll
  for (int kc = 0; kc < 16; ++kc) {
    v16h a = load_a_f32(arow + kc * 32);
#pragma unroll
    for (int u = 0; u < 4; ++u)
      acc[u] = wmma_f16(a, load_b_f16(brow0 + (size_t)u * 16 * 512 + kc * 32), acc[u]);
  }
#pragma unroll
  for (int u = 0; u < 4; ++u) {
    const int coln = ng * 64 + u * 16 + (lane & 15);
#pragma unroll
    for (int j = 0; j < 8; ++j) {
      int M = mt * 16 + j + lh * 8;
      int bb = M >> 14, l = M & 16383, w, t;
      tok2win(l, w, t);
      if (coln < 512) {
        int h = coln >> 6, d = coln & 63;
        kw[((((size_t)bb * 64 + w) * 8 + h) * 256 + t) * 64 + d] = (_Float16)acc[u][j];
      } else {
        int cc = coln - 512, h = cc >> 6, d = cc & 63;
        vt[((((size_t)bb * 64 + w) * 8 + h) * 64 + d) * 256 + t] = (_Float16)acc[u][j];
      }
    }
  }
}

// ---------------------------------------------------------------------------
// Attention: block = (b, window, head, 4 query-blocks), 4 waves.
// K tile (256x64 f16) staged into LDS once per block via the Tensor Data Mover
// with D# padding (row stride 144 B -> conflict-free b128 LDS reads).
// LDS region is reused for the P C-frag -> A-frag transpose after S phase.
// ---------------------------------------------------------------------------
__global__ void __launch_bounds__(128)
k_attn(const _Float16* __restrict__ qw, const _Float16* __restrict__ kw,
       const _Float16* __restrict__ vt, _Float16* __restrict__ ob) {
  __shared__ union {
    _Float16 k[256 * 72];              // K tile, 72 f16 (144 B) row stride
    _Float16 p[4][16][264];            // P staging, padded rows
  } sh;
  const int warp = threadIdx.x >> 5, lane = threadIdx.x & 31, lh = lane >> 4;
  const int wid = blockIdx.x * 4 + warp;
  const int mb = wid & 15;
  const int h  = (wid >> 4) & 7;
  const int w  = (wid >> 7) & 63;
  const int b  = wid >> 13;            // 4 warps of a block share (b,w,h)
  const _Float16* qh = qw + (((size_t)b * 64 + w) * 8 + h) * 256 * 64;
  const _Float16* kh = kw + (((size_t)b * 64 + w) * 8 + h) * 256 * 64;
  const _Float16* vh = vt + (((size_t)b * 64 + w) * 8 + h) * 64 * 256;

#if HAVE_TDM
  if (threadIdx.x < 32) {
    // D# group0: count=1 | lds_addr | global_addr | type=2
    unsigned lds = (unsigned)(size_t)&sh.k[0];
    unsigned long long ga = (unsigned long long)(size_t)kh;
    u32x4 g0 = {1u, lds, (unsigned)ga,
                (unsigned)((ga >> 32) & 0x01FFFFFFu) | (2u << 30)};
    // D# group1: data_size=1(2B) | pad_enable | pad_interval=4(32dw) | pad_amount=3(4dw);
    // tensor_dim0=64, tensor_dim1=256, tile_dim0=64, tile_dim1=256, stride0=64
    i32x8 g1 = {(int)0x07110000u,
                (int)(64u << 16),      // tensor_dim0[15:0] in [31:16]
                (int)(256u << 16),     // tensor_dim1[15:0] in [31:16]
                (int)(64u << 16),      // tile_dim0 in [31:16]
                256,                   // tile_dim1
                64,                    // tensor_dim0_stride[31:0]
                0, 0};
    i32x4 gz4 = {0, 0, 0, 0};
    i32x8 gz8 = {0, 0, 0, 0, 0, 0, 0, 0};
    __builtin_amdgcn_tensor_load_to_lds(g0, g1, gz4, gz4, gz8, 0);
    __builtin_amdgcn_s_wait_tensorcnt(0);
  }
#else
  // fallback: cooperative copy with padded rows (16B chunks, 144B row stride)
  for (int idx = threadIdx.x; idx < 256 * 8; idx += 128) {
    int row = idx >> 3, c8 = (idx & 7) * 8;
    *(v8h*)&sh.k[row * 72 + c8] = *(const v8h*)&kh[row * 64 + c8];
  }
#endif
  __syncthreads();

  // Q fragments: head-dim K=64 as two 32-chunks
  const _Float16* qrow = qh + (mb * 16 + (lane & 15)) * 64 + lh * 16;
  v16h aq0 = load_a_f16(qrow);
  v16h aq1 = load_a_f16(qrow + 32);

  // S = Q K^T over 16 key tiles; B operand = K rows from padded LDS
  v8f s[16];
#pragma unroll
  for (int ntt = 0; ntt < 16; ++ntt) {
    const _Float16* krow = &sh.k[(ntt * 16 + (lane & 15)) * 72 + lh * 16];
    v8f acc = {};
    acc    = wmma_f16(aq0, load_b_f16(krow),      acc);
    s[ntt] = wmma_f16(aq1, load_b_f16(krow + 32), acc);
  }

  // softmax over keys (keys: 16 lanes x 16 tiles; rows = vgprs j + 8*lh)
  const float scale = 0.125f;          // 64^-0.5
  float mx[8], sm[8];
#pragma unroll
  for (int j = 0; j < 8; ++j) mx[j] = -3.0e38f;
#pragma unroll
  for (int ntt = 0; ntt < 16; ++ntt)
#pragma unroll
    for (int j = 0; j < 8; ++j) {
      s[ntt][j] *= scale;
      mx[j] = fmaxf(mx[j], s[ntt][j]);
    }
#pragma unroll
  for (int off = 8; off; off >>= 1)
#pragma unroll
    for (int j = 0; j < 8; ++j) mx[j] = fmaxf(mx[j], __shfl_xor(mx[j], off, 32));
#pragma unroll
  for (int j = 0; j < 8; ++j) sm[j] = 0.0f;
#pragma unroll
  for (int ntt = 0; ntt < 16; ++ntt)
#pragma unroll
    for (int j = 0; j < 8; ++j) {
      float e = __expf(s[ntt][j] - mx[j]);
      s[ntt][j] = e;
      sm[j] += e;
    }
#pragma unroll
  for (int off = 8; off; off >>= 1)
#pragma unroll
    for (int j = 0; j < 8; ++j) sm[j] += __shfl_xor(sm[j], off, 32);
#pragma unroll
  for (int j = 0; j < 8; ++j) sm[j] = 1.0f / sm[j];

  __syncthreads();                     // all waves done reading K tile
  // stage P (f16) row-major [query][key] for the C-frag -> A-frag transpose
#pragma unroll
  for (int ntt = 0; ntt < 16; ++ntt)
#pragma unroll
    for (int j = 0; j < 8; ++j)
      sh.p[warp][j + lh * 8][ntt * 16 + (lane & 15)] = (_Float16)(s[ntt][j] * sm[j]);
  __syncthreads();

  // O = P V: N=64 dims (4 tiles), K=256 keys (8 chunks of 32)
  v8f o[4] = {};
#pragma unroll
  for (int kc = 0; kc < 8; ++kc) {
    v16h ap = load_a_f16(&sh.p[warp][lane & 15][kc * 32 + lh * 16]);
#pragma unroll
    for (int dt = 0; dt < 4; ++dt) {
      const _Float16* vrow = vh + (dt * 16 + (lane & 15)) * 256 + kc * 32 + lh * 16;
      o[dt] = wmma_f16(ap, load_b_f16(vrow), o[dt]);
    }
  }

  // de-window, store o as f16 [b][l][c]
  const int i1 = w >> 3, i3 = w & 7;
#pragma unroll
  for (int dt = 0; dt < 4; ++dt) {
    int cidx = h * 64 + dt * 16 + (lane & 15);
#pragma unroll
    for (int j = 0; j < 8; ++j) {
      int t = mb * 16 + j + lh * 8;
      int l = i1 * 2048 + (t >> 4) * 128 + i3 * 16 + (t & 15);
      ob[((size_t)b * 16384 + l) * 512 + cidx] = (_Float16)o[dt][j];
    }
  }
}

// ---------------------------------------------------------------------------
// out = o @ Wfc (f32 output). M=65536, N=512, K=512; 16x64 strip per wave
// ---------------------------------------------------------------------------
__global__ void __launch_bounds__(128)
k_fc(const _Float16* __restrict__ ob, const _Float16* __restrict__ WT,
     float* __restrict__ out) {
  const int wid  = blockIdx.x * (blockDim.x >> 5) + (threadIdx.x >> 5);
  const int lane = threadIdx.x & 31;
  const int lh   = lane >> 4;
  const int ng   = wid & 7;
  const int mt   = wid >> 3;
  const _Float16* arow  = ob + (size_t)(mt * 16 + (lane & 15)) * 512 + lh * 16;
  const _Float16* brow0 = WT + (size_t)(ng * 64 + (lane & 15)) * 512 + lh * 16;
  v8f acc[4] = {};
#pragma unroll
  for (int kc = 0; kc < 16; ++kc) {
    v16h a = load_a_f16(arow + kc * 32);
#pragma unroll
    for (int u = 0; u < 4; ++u)
      acc[u] = wmma_f16(a, load_b_f16(brow0 + (size_t)u * 16 * 512 + kc * 32), acc[u]);
  }
#pragma unroll
  for (int u = 0; u < 4; ++u) {
    const int coln = ng * 64 + u * 16 + (lane & 15);
#pragma unroll
    for (int j = 0; j < 8; ++j) {
      int M = mt * 16 + j + lh * 8;
      out[(size_t)M * 512 + coln] = acc[u][j];
    }
  }
}

// ---------------------------------------------------------------------------
extern "C" void kernel_launch(void* const* d_in, const int* in_sizes, int n_in,
                              void* d_out, int out_size, void* d_ws, size_t ws_size,
                              hipStream_t stream) {
  (void)in_sizes; (void)n_in; (void)out_size; (void)ws_size;
  const float* q   = (const float*)d_in[0];
  const float* kv  = (const float*)d_in[1];
  const float* Wq  = (const float*)d_in[2];
  const float* Wkv = (const float*)d_in[3];
  const float* Wfc = (const float*)d_in[4];
  float* out = (float*)d_out;

  char* ws = (char*)d_ws;
  const size_t SZ = (size_t)4 * 64 * 8 * 256 * 64;   // 33,554,432 f16 elements
  _Float16* qw   = (_Float16*)ws;  ws += SZ * 2;
  _Float16* kw   = (_Float16*)ws;  ws += SZ * 2;
  _Float16* vt   = (_Float16*)ws;  ws += SZ * 2;
  _Float16* ob   = (_Float16*)ws;  ws += SZ * 2;
  _Float16* WqT  = (_Float16*)ws;  ws += (size_t)512 * 512 * 2;
  _Float16* WkvT = (_Float16*)ws;  ws += (size_t)1024 * 512 * 2;
  _Float16* WfcT = (_Float16*)ws;

  hipLaunchKernelGGL(k_wtrans, dim3(1024), dim3(256), 0, stream, Wq,  WqT,  512,  512);
  hipLaunchKernelGGL(k_wtrans, dim3(2048), dim3(256), 0, stream, Wkv, WkvT, 512, 1024);
  hipLaunchKernelGGL(k_wtrans, dim3(1024), dim3(256), 0, stream, Wfc, WfcT, 512,  512);

  // q-proj: 4096 M-tiles x 8 strip-groups = 32768 waves / 4 per block
  hipLaunchKernelGGL(k_proj_q,  dim3(8192),  dim3(128), 0, stream, q,  WqT,  qw);
  // kv-proj: 4096 x 16 = 65536 waves / 4 per block
  hipLaunchKernelGGL(k_proj_kv, dim3(16384), dim3(128), 0, stream, kv, WkvT, kw, vt);
  // attention: 4*64*8*16 = 32768 waves / 4 per block
  hipLaunchKernelGGL(k_attn,    dim3(8192),  dim3(128), 0, stream, qw, kw, vt, ob);
  // final projection: 32768 waves / 4 per block
  hipLaunchKernelGGL(k_fc,      dim3(8192),  dim3(128), 0, stream, ob, WfcT, out);
}
